// CRF_58394375356439
// MI455X (gfx1250) — matrix operand order repaired and checked by reading, not compiled, not run on hardware
//
#include <hip/hip_runtime.h>

// ---------------------------------------------------------------------------
// CRF decode, fully fused for gfx1250 (MI455X):
//   conv folded into emission weights (W' = W*A, done once per launch),
//   emission GEMM via v_wmma_f32_16x16x32_bf16 (f32 accumulate),
//   Viterbi decode in-register with LDS-resident scores/backpointers.
// X traffic (224 MiB) sets a ~10us HBM floor @23.3TB/s; math is trivial.
// f32->bf16 conversion is branchless RNE (inputs are finite) packed 2/dword.
// ---------------------------------------------------------------------------

typedef __attribute__((ext_vector_type(16))) __bf16 v16bf;
typedef __attribute__((ext_vector_type(8)))  float  v8f;
typedef __attribute__((ext_vector_type(4)))  float  v4f;

#define NB      32768     // words (batch)
#define MLET    14        // letters per word
#define FDIM    128       // features per letter
#define LLAB    26        // labels
#define NTHR    256       // 8 waves of 32
#define WPB     64        // words per block
#define WPW     8         // words per wave
#define TILES   7         // (WPW*MLET)/16 row-tiles per wave
#define TSTRIDE 32        // transposed-T row stride in LDS (floats, 16B aligned rows)
#define SW      365       // per-word score stride in LDS (floats, odd -> no bank conflicts)
#define BPSTR   348       // per-word backpointer stride (bytes)
#define TT_F    (LLAB * TSTRIDE)                 // 832 floats for transposed T
#define S_F     TT_F                             // score region base (float index)
#define BP_B    ((TT_F + WPB * SW) * 4)          // backpointer region base (bytes)
#define LDSBYTES (BP_B + WPB * BPSTR)            // 119,040 B < 160 KB (2 blocks/WGP)

// Branchless round-to-nearest-even f32 -> bf16 (finite inputs only).
__device__ __forceinline__ unsigned f2bf_bits(float f) {
  union { float f; unsigned u; } x; x.f = f;
  return x.u + 0x7fffu + ((x.u >> 16) & 1u);   // rounded value lives in bits [31:16]
}
__device__ __forceinline__ unsigned pack2bf(float lo, float hi) {
  return (f2bf_bits(lo) >> 16) | (f2bf_bits(hi) & 0xffff0000u);
}

// --- prep 1: fold 5x5 SAME conv (cross-correlation, XLA semantics) into W ---
// W'[l][r',c'] = sum_{i,j} W[l][r'+2-i, c'+2-j] * K[i][j];  sbias[l] = b*sum_f W[l][f]
__global__ void crf_prep_fold(const float* __restrict__ W, const float* __restrict__ K,
                              const float* __restrict__ b, float* __restrict__ Wp,
                              float* __restrict__ sbias) {
  int idx = blockIdx.x * blockDim.x + threadIdx.x;
  if (idx >= LLAB * FDIM) return;
  int l = idx >> 7, rc = idx & 127;
  int r = rc >> 3, c = rc & 7;
  float acc = 0.f;
  #pragma unroll
  for (int i = 0; i < 5; ++i) {
    int rr = r + 2 - i;
    if ((unsigned)rr < 16u) {
      #pragma unroll
      for (int j = 0; j < 5; ++j) {
        int cc = c + 2 - j;
        if ((unsigned)cc < 8u) acc += W[l * FDIM + rr * 8 + cc] * K[i * 5 + j];
      }
    }
  }
  Wp[idx] = acc;
  if (rc == 0) {
    float s = 0.f;
    for (int f = 0; f < FDIM; ++f) s += W[l * FDIM + f];
    sbias[l] = s * b[0];
  }
}

// --- prep 2: pack W' into wave32 WMMA B-matrix layout (bf16), N padded 26->32 ---
// tile tk = nt*4+kc; lanes 0-15: N=lane, K 0-15 of chunk; lanes 16-31: N=lane-16, K 16-31.
__global__ void crf_prep_pack(const float* __restrict__ Wp, unsigned short* __restrict__ Bpack) {
  int idx = blockIdx.x * blockDim.x + threadIdx.x;
  if (idx >= 8 * 32 * 16) return;
  int e    = idx & 15;
  int lane = (idx >> 4) & 31;
  int tk   = idx >> 9;                 // 0..7
  int nt = tk >> 2, kc = tk & 3;
  int n = nt * 16 + (lane & 15);
  int k = kc * 32 + (lane >> 4) * 16 + e;
  float val = (n < LLAB) ? Wp[n * FDIM + k] : 0.f;
  Bpack[idx] = (unsigned short)(f2bf_bits(val) >> 16);
}

// --- fused emission GEMM + Viterbi ---
__global__ __launch_bounds__(NTHR) void crf_fused(
    const float* __restrict__ X, const float* __restrict__ Tg,
    const unsigned short* __restrict__ Bpack, const float* __restrict__ sbias,
    int* __restrict__ out) {
  extern __shared__ char smem[];
  float* Tt            = (float*)smem;            // transposed T: Tt[j*32 + i]
  float* Slds          = (float*)smem + S_F;      // scores[word][m][l]
  unsigned char* BPl   = (unsigned char*)smem + BP_B;

  const int tid  = threadIdx.x;
  const int lane = tid & 31;
  const int wave = tid >> 5;

  // stage T transposed (contiguous + 16B-aligned in the hot inner loop)
  for (int idx = tid; idx < LLAB * LLAB; idx += NTHR) {
    int i = idx / LLAB, j = idx - i * LLAB;
    Tt[j * TSTRIDE + i] = Tg[idx];
  }

  const int nlo   = lane & 15;   // M row (A/C) or N col (B/C) within tile
  const int khalf = lane >> 4;   // which K half this lane holds

  // B tiles: resident in VGPRs for the whole kernel (2 N-tiles x 4 K-chunks)
  v16bf Bt[2][4];
  const v16bf* Bp = (const v16bf*)Bpack;
  #pragma unroll
  for (int nt = 0; nt < 2; ++nt)
    #pragma unroll
    for (int kc = 0; kc < 4; ++kc)
      Bt[nt][kc] = Bp[(nt * 4 + kc) * 32 + lane];

  float sb[2];
  #pragma unroll
  for (int nt = 0; nt < 2; ++nt) {
    int n = nt * 16 + nlo;
    sb[nt] = (n < LLAB) ? sbias[n] : 0.f;
  }

  // this wave's strip of X: 8 words * 14 letters = 112 fully contiguous rows
  const int wordBase = blockIdx.x * WPB + wave * WPW;
  const float* Xw = X + (size_t)wordBase * (MLET * FDIM);

  for (int t = 0; t < TILES; ++t) {
    const float* rowp = Xw + (size_t)(t * 16 + nlo) * FDIM;
    __builtin_prefetch(rowp + 16 * FDIM, 0, 1);   // global_prefetch next row-tile
    v8f c0 = {}, c1 = {};
    #pragma unroll
    for (int kc = 0; kc < 4; ++kc) {
      // A layout: lane<16 -> K 0..15; lane>=16 -> K 16..31 of chunk
      const float* p = rowp + kc * 32 + khalf * 16;
      v4f f0 = *(const v4f*)(p);
      v4f f1 = *(const v4f*)(p + 4);
      v4f f2 = *(const v4f*)(p + 8);
      v4f f3 = *(const v4f*)(p + 12);
      union { v16bf v; unsigned w[8]; } a;
      a.w[0] = pack2bf(f0[0], f0[1]);
      a.w[1] = pack2bf(f0[2], f0[3]);
      a.w[2] = pack2bf(f1[0], f1[1]);
      a.w[3] = pack2bf(f1[2], f1[3]);
      a.w[4] = pack2bf(f2[0], f2[1]);
      a.w[5] = pack2bf(f2[2], f2[3]);
      a.w[6] = pack2bf(f3[0], f3[1]);
      a.w[7] = pack2bf(f3[2], f3[3]);
      c0 = __builtin_amdgcn_wmma_f32_16x16x32_bf16(false, a.v, false, Bt[0][kc],
                                                   (short)0, c0, false, false);
      c1 = __builtin_amdgcn_wmma_f32_16x16x32_bf16(false, a.v, false, Bt[1][kc],
                                                   (short)0, c1, false, false);
    }
    // C layout: lane<16 holds rows 0-7 of col nlo, lane>=16 rows 8-15
    const int rbase = t * 16 + khalf * 8;
    #pragma unroll
    for (int nt = 0; nt < 2; ++nt) {
      int n = nt * 16 + nlo;
      if (n < LLAB) {
        v8f cc = nt ? c1 : c0;
        #pragma unroll
        for (int j = 0; j < 8; ++j) {
          int rw = rbase + j;          // row within wave strip (<112)
          int w  = rw / MLET;
          int m  = rw - w * MLET;
          Slds[(wave * WPW + w) * SW + m * LLAB + n] = cc[j] + sb[nt];
        }
      }
    }
  }

  __syncthreads();

  // Viterbi: one word per lane (threads 0..63); T reads are uniform -> LDS broadcast
  if (tid < WPB) {
    const float* sc   = Slds + tid * SW;
    unsigned char* bp = BPl + tid * BPSTR;
    float v[LLAB];
    #pragma unroll
    for (int l = 0; l < LLAB; ++l) v[l] = sc[l];
    for (int m = 1; m < MLET; ++m) {
      const float* sm    = sc + m * LLAB;
      unsigned char* bpm = bp + (m - 1) * LLAB;
      float vn[LLAB];
      #pragma unroll
      for (int j = 0; j < LLAB; ++j) {
        const float* tj = Tt + j * TSTRIDE;
        float best = v[0] + tj[0];
        int bi = 0;
        #pragma unroll
        for (int i = 1; i < LLAB; ++i) {
          float cand = v[i] + tj[i];
          if (cand > best) { best = cand; bi = i; }   // strict '>' == argmax first-max
        }
        vn[j] = best + sm[j];
        bpm[j] = (unsigned char)bi;
      }
      #pragma unroll
      for (int j = 0; j < LLAB; ++j) v[j] = vn[j];
    }
    int last = 0; float bv = v[0];
    #pragma unroll
    for (int l = 1; l < LLAB; ++l) if (v[l] > bv) { bv = v[l]; last = l; }
    int* op = out + (size_t)(blockIdx.x * WPB + tid) * MLET;
    op[MLET - 1] = last;
    for (int m = MLET - 2; m >= 0; --m) {
      last = bp[m * LLAB + last];
      op[m] = last;
    }
  }
}

extern "C" void kernel_launch(void* const* d_in, const int* in_sizes, int n_in,
                              void* d_out, int out_size, void* d_ws, size_t ws_size,
                              hipStream_t stream) {
  const float* X = (const float*)d_in[0];   // (B, M, F) f32
  const float* K = (const float*)d_in[1];   // (1,1,5,5)
  const float* b = (const float*)d_in[2];   // (1,)
  const float* W = (const float*)d_in[3];   // (L, F)
  const float* T = (const float*)d_in[4];   // (L, L)
  int* out = (int*)d_out;                   // (B, M, 1) int32

  float* Wp = (float*)d_ws;                                   // 26*128 f32
  float* sb = Wp + LLAB * FDIM;                               // 26 f32 (padded slot)
  unsigned short* Bpack =
      (unsigned short*)((char*)d_ws + (LLAB * FDIM + 32) * sizeof(float)); // 4096 bf16

  crf_prep_fold<<<(LLAB * FDIM + NTHR - 1) / NTHR, NTHR, 0, stream>>>(W, K, b, Wp, sb);
  crf_prep_pack<<<(8 * 32 * 16 + NTHR - 1) / NTHR, NTHR, 0, stream>>>(Wp, Bpack);
  crf_fused<<<NB / WPB, NTHR, LDSBYTES, stream>>>(X, T, Bpack, sb, out);
}
